// tf_d_1_39316130628068
// MI455X (gfx1250) — matrix operand order repaired and checked by reading, not compiled, not run on hardware
//
#include <hip/hip_runtime.h>
#include <hip/hip_bf16.h>

typedef __attribute__((ext_vector_type(16))) _Float16 v16h;
typedef __attribute__((ext_vector_type(8)))  float    v8f;

union AFrag { v16h v; unsigned int u[8]; };
union HPack { _Float16 h[8]; unsigned int u[4]; };

__device__ inline unsigned short f2h(float x) {
  _Float16 h = (_Float16)x;
  unsigned short u;
  __builtin_memcpy(&u, &h, 2);
  return u;
}

__device__ inline float sigmoidf_(float x) { return 1.f / (1.f + __expf(-x)); }
__device__ inline float tanh_(float x) {
  x = fminf(fmaxf(x, -15.f), 15.f);
  float e = __expf(2.f * x);
  return (e - 1.f) / (e + 1.f);
}

// ---------------------------------------------------------------------------
// Weight prep: f32 -> f16 (padded), fused LSTM bias
// ---------------------------------------------------------------------------
__global__ void __launch_bounds__(256) prep_kernel(
    const float* __restrict__ W_ih, const float* __restrict__ W_hh,
    const float* __restrict__ b_ih, const float* __restrict__ b_hh,
    const float* __restrict__ W1,   const float* __restrict__ b1,
    const float* __restrict__ W2,   const float* __restrict__ W3,
    unsigned short* __restrict__ wih16, unsigned short* __restrict__ whh16,
    float* __restrict__ biasg, unsigned short* __restrict__ w1f,
    float* __restrict__ b1p, unsigned short* __restrict__ w2f,
    unsigned short* __restrict__ w3f)
{
  const int tid = blockIdx.x * blockDim.x + threadIdx.x;
  const int stride = gridDim.x * blockDim.x;
  for (int i = tid; i < 512 * 576; i += stride) {          // W_ih: [512][551] -> [512][576]
    int g = i / 576, k = i % 576;
    wih16[i] = f2h(k < 551 ? W_ih[g * 551 + k] : 0.f);
  }
  for (int i = tid; i < 512 * 128; i += stride) whh16[i] = f2h(W_hh[i]);
  for (int i = tid; i < 512; i += stride) biasg[i] = b_ih[i] + b_hh[i];
  for (int i = tid; i < 640 * 640; i += stride) {          // W1: [531][640] -> [640][640]
    int n = i / 640;
    w1f[i] = f2h(n < 531 ? W1[i] : 0.f);
  }
  for (int i = tid; i < 640; i += stride) b1p[i] = (i < 531) ? b1[i] : 0.f;
  for (int i = tid; i < 256 * 640; i += stride) {          // W2: [256][531] -> [256][640]
    int n = i / 640, k = i % 640;
    w2f[i] = f2h(k < 531 ? W2[n * 531 + k] : 0.f);
  }
  for (int i = tid; i < 64 * 256; i += stride) w3f[i] = f2h(W3[i]);
}

// ---------------------------------------------------------------------------
// Generic WMMA GEMM: C[M][ldc] = act( A(f32)[M][K] * Bw(f16)[N][K]^T + bias )
// mode 1: A is concat(x1[.,300], x2[.,251], zeros) with K=576 (branchless).
// Block: 256 threads = 8 waves (4M x 2N), block tile 64M x 128N, K-step 32.
// Double-buffered LDS software pipeline; 16B-aligned LDS rows (stride 40).
// ---------------------------------------------------------------------------
#define STRA 40
#define STRB 40
__global__ void __launch_bounds__(256) gemm_f16_relu_kernel(
    const float* __restrict__ A, const float* __restrict__ X1,
    const float* __restrict__ X2,
    const unsigned short* __restrict__ Bw, const float* __restrict__ bias,
    float* __restrict__ C, int M, int N, int K, int ldc, int NB,
    int mode, int relu)
{
  __shared__ __align__(16) unsigned short shA[2][64 * STRA];
  __shared__ __align__(16) unsigned short shB[2][128 * STRB];

  const int tid  = threadIdx.x;
  const int lane = tid & 31;
  const int wid  = tid >> 5;
  const int l16  = lane & 15;
  const int half = lane >> 4;
  const int waveM = wid >> 1;
  const int waveN = wid & 1;
  const int mBase = blockIdx.x * 64;
  const int nBase = blockIdx.y * 128;
  const int nk = K >> 5;

  v8f acc[4];
#pragma unroll
  for (int c = 0; c < 4; c++)
#pragma unroll
    for (int r = 0; r < 8; r++) acc[c][r] = 0.f;

  // staging assignments
  const int arow = tid >> 2;        // 0..63
  const int aseg = (tid & 3) * 8;   // 0,8,16,24
  const int brow = tid >> 1;        // 0..127
  const int bseg = (tid & 1) * 16;  // 0,16
  const int gm = mBase + arow;
  int gn = nBase + brow;
  if (gn >= NB) gn = 0;             // clamp; masked at store
  const unsigned short* bsrc = Bw + (size_t)gn * K + bseg;

  HPack pa;        // incoming A tile fragment (8 f16)
  uint4 pb0, pb1;  // incoming B tile fragment (16 f16)

  // ---- fetch tile ks into registers (branchless concat) ----
  auto fetch = [&](int ks) {
    const int kb = ks * 32;
    if (mode == 0) {
      const float* ap = A + (size_t)gm * K + kb + aseg;
#pragma unroll
      for (int j = 0; j < 8; j++) pa.h[j] = (_Float16)ap[j];
      __builtin_prefetch(ap + 64, 0, 0);   // two k-tiles ahead (HBM stream)
    } else {
#pragma unroll
      for (int j = 0; j < 8; j++) {
        const int gk = kb + aseg + j;
        const int gkc = (gk < 551) ? gk : 550;
        const float* p = (gkc < 300) ? (X1 + (size_t)gm * 300 + gkc)
                                     : (X2 + (size_t)gm * 251 + (gkc - 300));
        float v = *p;                       // single predicated-address load
        pa.h[j] = (_Float16)((gk < 551) ? v : 0.f);
      }
    }
    const uint4* bs = (const uint4*)(bsrc + kb);
    pb0 = bs[0];
    pb1 = bs[1];
  };
  auto stash = [&](int ks) {
    const int buf = ks & 1;
    uint4* da = (uint4*)&shA[buf][arow * STRA + aseg];
    uint4 qa; qa.x = pa.u[0]; qa.y = pa.u[1]; qa.z = pa.u[2]; qa.w = pa.u[3];
    da[0] = qa;
    uint4* db = (uint4*)&shB[buf][brow * STRB + bseg];
    db[0] = pb0;
    db[1] = pb1;
  };

  fetch(0);
  stash(0);

  for (int ks = 0; ks < nk; ks++) {
    __syncthreads();                 // tile ks visible; tile ks-2 reads done
    const int cur = ks & 1;
    const bool more = (ks + 1) < nk;
    if (more) fetch(ks + 1);         // global loads overlap WMMA below

    // A fragment: 16x32 f16, pair K = 2j + 16*half
    AFrag af;
    {
      const unsigned short* ap = &shA[cur][(waveM * 16 + l16) * STRA];
#pragma unroll
      for (int j = 0; j < 8; j++) {
        const int k0 = 2 * j + 16 * half;
        af.u[j] = *(const unsigned int*)&ap[k0];
      }
    }
#pragma unroll
    for (int c = 0; c < 4; c++) {
      const uint4* bp =
          (const uint4*)&shB[cur][(waveN * 64 + c * 16 + l16) * STRB + half * 16];
      uint4 q0 = bp[0], q1 = bp[1];
      AFrag bf;  // element e -> K = e + 16*half, N = l16
      bf.u[0] = q0.x; bf.u[1] = q0.y; bf.u[2] = q0.z; bf.u[3] = q0.w;
      bf.u[4] = q1.x; bf.u[5] = q1.y; bf.u[6] = q1.z; bf.u[7] = q1.w;
      acc[c] = __builtin_amdgcn_wmma_f32_16x16x32_f16(
          false, af.v, false, bf.v, (short)0, acc[c], false, false);
    }
    if (more) stash(ks + 1);
  }

  // epilogue: bias + optional relu, D layout M = r + 8*half, N = l16
#pragma unroll
  for (int c = 0; c < 4; c++) {
    const int gnn = nBase + waveN * 64 + c * 16 + l16;
    const bool ok = (gnn < N);
    const float bv = (ok && bias) ? bias[gnn] : 0.f;
#pragma unroll
    for (int r = 0; r < 8; r++) {
      const int gmm = mBase + waveM * 16 + r + 8 * half;
      float v = acc[c][r] + bv;
      if (relu) v = fmaxf(v, 0.f);
      if (ok) C[(size_t)gmm * ldc + gnn] = v;
    }
  }
}

// ---------------------------------------------------------------------------
// LSTM: 64 batch rows / block, h_{t-1} kept as f16 in LDS, c in VGPRs.
// xg already contains x@W_ih^T + b_ih + b_hh. Gate order i,f,g,o.
// ---------------------------------------------------------------------------
__global__ void __launch_bounds__(256) lstm_kernel(
    const float* __restrict__ xg,             // [B*5][512]
    const unsigned short* __restrict__ whh,   // [512][128] f16
    float* __restrict__ h_all)                // [B][640]  (t*128 + j)
{
  __shared__ unsigned short shH[64 * 130];                  // h f16
  __shared__ __align__(16) unsigned short shB[512 * STRB];  // W_hh k-slice

  const int tid  = threadIdx.x;
  const int lane = tid & 31;
  const int wid  = tid >> 5;
  const int l16  = lane & 15;
  const int half = lane >> 4;
  const int waveM = wid >> 1;   // row group 0..3
  const int waveN = wid & 1;    // hidden-colgroup half
  const int bBase = blockIdx.x * 64;

  for (int i = tid; i < 64 * 130; i += 256) shH[i] = 0;   // h_{-1} = 0

  v8f cst[4];
#pragma unroll
  for (int c = 0; c < 4; c++)
#pragma unroll
    for (int r = 0; r < 8; r++) cst[c][r] = 0.f;

  for (int t = 0; t < 5; t++) {
    v8f acc[4][4];  // [hidden colgroup][gate]
    // init from xg: one base per r, constant instruction offsets g*128+cg*16
#pragma unroll
    for (int r = 0; r < 8; r++) {
      const int row = bBase + waveM * 16 + r + 8 * half;
      const float* base = xg + ((size_t)row * 5 + t) * 512 + waveN * 64 + l16;
#pragma unroll
      for (int cg = 0; cg < 4; cg++)
#pragma unroll
        for (int g = 0; g < 4; g++)
          acc[cg][g][r] = base[g * 128 + cg * 16];
    }
    // acc += h_{t-1} @ W_hh^T
    for (int ks = 0; ks < 4; ks++) {
      __syncthreads();
#pragma unroll
      for (int nn = 0; nn < 2; nn++) {   // stage 512x32 W_hh slice (b128 copies)
        const int n = tid * 2 + nn;
        const uint4* src = (const uint4*)(whh + (size_t)n * 128 + ks * 32);
        uint4* dst = (uint4*)&shB[n * STRB];
        dst[0] = src[0]; dst[1] = src[1]; dst[2] = src[2]; dst[3] = src[3];
      }
      __syncthreads();
      AFrag af;
      const int ar = waveM * 16 + l16;
#pragma unroll
      for (int j = 0; j < 8; j++) {
        const int k0 = ks * 32 + 2 * j + 16 * half;
        af.u[j] = *(const unsigned int*)&shH[ar * 130 + k0];
      }
#pragma unroll
      for (int cg = 0; cg < 4; cg++) {
        const int gcg = waveN * 4 + cg;
#pragma unroll
        for (int g = 0; g < 4; g++) {
          const uint4* bp =
              (const uint4*)&shB[(g * 128 + gcg * 16 + l16) * STRB + half * 16];
          uint4 q0 = bp[0], q1 = bp[1];
          AFrag bf;
          bf.u[0] = q0.x; bf.u[1] = q0.y; bf.u[2] = q0.z; bf.u[3] = q0.w;
          bf.u[4] = q1.x; bf.u[5] = q1.y; bf.u[6] = q1.z; bf.u[7] = q1.w;
          acc[cg][g] = __builtin_amdgcn_wmma_f32_16x16x32_f16(
              false, af.v, false, bf.v, (short)0, acc[cg][g], false, false);
        }
      }
    }
    __syncthreads();  // all shH reads done before overwrite

    // gates, state update, write h (LDS f16 + global f32)
#pragma unroll
    for (int cg = 0; cg < 4; cg++) {
      const int gcg = waveN * 4 + cg;
      const int col = gcg * 16 + l16;
#pragma unroll
      for (int r = 0; r < 8; r++) {
        const float iv = sigmoidf_(acc[cg][0][r]);
        const float fv = sigmoidf_(acc[cg][1][r]);
        const float gv = tanh_(acc[cg][2][r]);
        const float ov = sigmoidf_(acc[cg][3][r]);
        const float cv = fv * cst[cg][r] + iv * gv;
        cst[cg][r] = cv;
        const float hv = ov * tanh_(cv);
        const int rowL = waveM * 16 + r + 8 * half;
        shH[rowL * 130 + col] = f2h(hv);
        h_all[(size_t)(bBase + rowL) * 640 + t * 128 + col] = hv;
      }
    }
    __syncthreads();
  }
}

// ---------------------------------------------------------------------------
// FC4 (64 -> 2) + softmax
// ---------------------------------------------------------------------------
__global__ void __launch_bounds__(256) head_kernel(
    const float* __restrict__ act3, const float* __restrict__ W4,
    const float* __restrict__ b4, float* __restrict__ out, int Bn)
{
  const int b = blockIdx.x * 256 + threadIdx.x;
  if (b >= Bn) return;
  float l0 = b4[0], l1 = b4[1];
  const float* a = act3 + (size_t)b * 64;
#pragma unroll 8
  for (int k = 0; k < 64; k++) {
    const float av = a[k];
    l0 += av * W4[k];
    l1 += av * W4[64 + k];
  }
  const float m = fmaxf(l0, l1);
  const float e0 = __expf(l0 - m), e1 = __expf(l1 - m);
  const float inv = 1.f / (e0 + e1);
  out[(size_t)b * 2 + 0] = e0 * inv;
  out[(size_t)b * 2 + 1] = e1 * inv;
}

// ---------------------------------------------------------------------------
extern "C" void kernel_launch(void* const* d_in, const int* in_sizes, int n_in,
                              void* d_out, int out_size, void* d_ws, size_t ws_size,
                              hipStream_t stream)
{
  (void)in_sizes; (void)n_in; (void)out_size; (void)ws_size;
  const float* x1   = (const float*)d_in[0];
  const float* x2   = (const float*)d_in[1];
  const float* W_ih = (const float*)d_in[2];
  const float* W_hh = (const float*)d_in[3];
  const float* b_ih = (const float*)d_in[4];
  const float* b_hh = (const float*)d_in[5];
  const float* W1   = (const float*)d_in[6];
  const float* b1   = (const float*)d_in[7];
  const float* W2   = (const float*)d_in[8];
  const float* b2   = (const float*)d_in[9];
  const float* W3   = (const float*)d_in[10];
  const float* b3   = (const float*)d_in[11];
  const float* W4   = (const float*)d_in[12];
  const float* b4   = (const float*)d_in[13];

  char* ws = (char*)d_ws;
  size_t off = 0;
  auto take = [&](size_t bytes) -> char* {
    char* p = ws + off;
    off = (off + bytes + 255) & ~(size_t)255;
    return p;
  };

  unsigned short* wih16 = (unsigned short*)take((size_t)512 * 576 * 2);
  unsigned short* whh16 = (unsigned short*)take((size_t)512 * 128 * 2);
  float*          biasg = (float*)take(512 * 4);
  unsigned short* w1f   = (unsigned short*)take((size_t)640 * 640 * 2);
  float*          b1p   = (float*)take(640 * 4);
  unsigned short* w2f   = (unsigned short*)take((size_t)256 * 640 * 2);
  unsigned short* w3f   = (unsigned short*)take((size_t)64 * 256 * 2);
  float* xg    = (float*)take((size_t)81920 * 512 * 4);   // 168 MB
  float* h_all = (float*)take((size_t)16384 * 640 * 4);   //  42 MB
  float* act1  = (float*)take((size_t)16384 * 640 * 4);   //  42 MB
  float* act2  = (float*)take((size_t)16384 * 256 * 4);   //  17 MB
  float* act3  = (float*)take((size_t)16384 * 64 * 4);    //   4 MB

  prep_kernel<<<512, 256, 0, stream>>>(W_ih, W_hh, b_ih, b_hh, W1, b1, W2, W3,
                                       wih16, whh16, biasg, w1f, b1p, w2f, w3f);

  // xg = concat(x1,x2) @ W_ih^T + (b_ih + b_hh)   [81920 x 512]
  gemm_f16_relu_kernel<<<dim3(1280, 4), 256, 0, stream>>>(
      nullptr, x1, x2, wih16, biasg, xg, 81920, 512, 576, 512, 512, 1, 0);

  lstm_kernel<<<256, 256, 0, stream>>>(xg, whh16, h_all);

  // FC1: [16384 x 640] -> 640 (531 real, padded), relu
  gemm_f16_relu_kernel<<<dim3(256, 5), 256, 0, stream>>>(
      h_all, nullptr, nullptr, w1f, b1p, act1, 16384, 640, 640, 640, 640, 0, 1);

  // FC2: -> 256, relu  (K padded to 640; act1 pad cols are zero)
  gemm_f16_relu_kernel<<<dim3(256, 2), 256, 0, stream>>>(
      act1, nullptr, nullptr, w2f, b2, act2, 16384, 256, 640, 256, 256, 0, 1);

  // FC3: -> 64, relu
  gemm_f16_relu_kernel<<<dim3(256, 1), 256, 0, stream>>>(
      act2, nullptr, nullptr, w3f, b3, act3, 16384, 64, 256, 64, 64, 0, 1);

  head_kernel<<<64, 256, 0, stream>>>(act3, W4, b4, (float*)d_out, 16384);
}
